// InstaNas_24584392802527
// MI455X (gfx1250) — matrix operand order repaired and checked by reading, not compiled, not run on hardware
//
#include <hip/hip_runtime.h>
#include <hip/hip_bf16.h>
#include <stdint.h>

// ---------------------------------------------------------------------------
// Types for CDNA5 WMMA (wave32): 16x16x32 bf16 -> f32
// ---------------------------------------------------------------------------
typedef __attribute__((ext_vector_type(16))) __bf16 v16bf_t;
typedef __attribute__((ext_vector_type(8)))  float  v8f_t;

__device__ __forceinline__ float bfu2f(unsigned short s) {
  unsigned u = ((unsigned)s) << 16;
  return __builtin_bit_cast(float, u);
}
__device__ __forceinline__ unsigned short f2bfu(float f) {
  unsigned u = __builtin_bit_cast(unsigned, f);
  unsigned r = u + 0x7FFFu + ((u >> 16) & 1u);  // round-to-nearest-even
  return (unsigned short)(r >> 16);
}

enum { F_RELU = 1, F_BN = 2, F_RES = 4, F_MASKACC = 8 };

#define BM 128
#define BN 64
#define BK 64    // two 16x16x32 K-steps per staged tile
#define LDA 72   // padded LDS row stride (bf16): 144B rows, 16B aligned
#define LDB 72

// ---------------------------------------------------------------------------
// Double-buffered WMMA implicit-GEMM: out[M,N] = A[M,K](bf16) * W[N,K](f32)
// Pipeline per K-tile: issue async A(next) + B(next) global loads ->
//   8x v_wmma on current tile -> packed-store B(next) to LDS ->
//   s_wait_asynccnt -> barrier.
// Fused epilogue: BN -> +residual -> ReLU -> bf16 store or policy-masked
//   f32 accumulate.  NOTE: K is always even for every GEMM in this net,
//   so k-pairs are always 4B/8B aligned.
// ---------------------------------------------------------------------------
__launch_bounds__(256)
__global__ void wmma_gemm_bf16(const unsigned short* __restrict__ A,
                               const float* __restrict__ W,
                               float* __restrict__ outF,
                               unsigned short* __restrict__ outH,
                               const float* __restrict__ gamma,
                               const float* __restrict__ beta,
                               const unsigned short* __restrict__ residual,
                               const int* __restrict__ policy,
                               int t_idx, int act_idx, int T, int HW,
                               int M, int N, int K, int flags) {
  __shared__ __align__(16) unsigned short sA[2][BM * LDA];
  __shared__ __align__(16) unsigned short sB[2][BN * LDB];

  const int tid   = threadIdx.x;
  const int lane  = tid & 31;
  const int wave  = tid >> 5;
  const int waveM = wave & 3;   // 4 waves along M, 32 rows each
  const int waveN = wave >> 2;  // 2 waves along N, 32 cols each
  const int m0 = blockIdx.y * BM;
  const int n0 = blockIdx.x * BN;
  const int hl  = lane >> 4;    // lane half (0: lanes 0-15, 1: lanes 16-31)
  const int l16 = lane & 15;

  v8f_t acc[2][2];
#pragma unroll
  for (int mi = 0; mi < 2; ++mi)
#pragma unroll
    for (int ni = 0; ni < 2; ++ni) acc[mi][ni] = (v8f_t)(0.0f);

  const int nTiles = (K + BK - 1) / BK;

  // ---- A tile staging: async fast path (full tile) or clamped pair loads --
  auto stageA = [&](int buf, int k0) -> bool {
    const bool fullA = (m0 + BM <= M) && (k0 + BK <= K);
    if (fullA) {
      const unsigned base = (unsigned)(uintptr_t)(&sA[buf][0]);
      // 1024 x 16B chunks, 4 per thread, tracked by ASYNCcnt
#pragma unroll
      for (int it = 0; it < 4; ++it) {
        int chunk = tid + it * 256;
        int row = chunk >> 3;
        int seg = chunk & 7;
        unsigned ldsOff = base + (unsigned)(row * LDA + seg * 8) * 2u;
        unsigned gOff =
            (unsigned)(((size_t)(m0 + row) * (size_t)K + (size_t)k0 + seg * 8) * 2u);
        asm volatile("global_load_async_to_lds_b128 %0, %1, %2"
                     :: "v"(ldsOff), "v"(gOff), "s"(A) : "memory");
      }
    } else {
      // clamped u32 (bf16 pair) loads, b32 LDS stores, zero via select
#pragma unroll
      for (int it = 0; it < 16; ++it) {
        int idx = tid + it * 256;      // pair index
        int row = idx >> 5;            // 32 pairs per 64-wide row
        int kp  = (idx & 31) * 2;
        int gm = m0 + row, gk = k0 + kp;
        int cm = gm < M ? gm : M - 1;
        int ck = gk < K ? gk : 0;      // K even => gk even, gk<K => gk+1<K
        unsigned v = *(const unsigned*)(A + (size_t)cm * K + ck);
        if (gm >= M || gk >= K) v = 0u;
        *(unsigned*)(&sA[buf][row * LDA + kp]) = v;
      }
    }
    return fullA;
  };

  // ---- B tile: f32 weight pairs -> regs (merged b64 loads), pack later ----
  auto loadB = [&](int k0, float* breg) {
#pragma unroll
    for (int it = 0; it < 8; ++it) {
      int idx = tid + it * 256;        // pair index
      int n  = idx >> 5;               // 32 pairs per 64-wide row
      int kp = (idx & 31) * 2;
      int gn = n0 + n, gk = k0 + kp;
      int cn = gn < N ? gn : N - 1;
      int ck = gk < K ? gk : 0;
      const float* wp = W + (size_t)cn * K + ck;   // 8B aligned (K even)
      float a = wp[0];
      float b = wp[1];
      if (gn >= N || gk >= K) { a = 0.0f; b = 0.0f; }
      breg[2 * it]     = a;
      breg[2 * it + 1] = b;
    }
  };
  auto storeB = [&](int buf, const float* breg) {
#pragma unroll
    for (int it = 0; it < 8; ++it) {
      int idx = tid + it * 256;
      int n  = idx >> 5;
      int kp = (idx & 31) * 2;
      unsigned pk = (unsigned)f2bfu(breg[2 * it]) |
                    ((unsigned)f2bfu(breg[2 * it + 1]) << 16);
      *(unsigned*)(&sB[buf][n * LDB + kp]) = pk;   // packed ds_store_b32
    }
  };

  // ---- compute: 2 K-steps x (2x2) fragments per staged tile ----
  auto compute = [&](int buf) {
#pragma unroll
    for (int ks = 0; ks < 2; ++ks) {
      union Frag { unsigned u[8]; v16bf_t v; } af[2], bw[2];
#pragma unroll
      for (int mi = 0; mi < 2; ++mi) {
        int mrow = waveM * 32 + mi * 16 + l16;
        const unsigned short* ap = &sA[buf][mrow * LDA + ks * 32];
#pragma unroll
        for (int q = 0; q < 8; ++q) {
          int kb = 2 * q + hl * 16;
          af[mi].u[q] = *(const unsigned*)(ap + kb);
        }
      }
#pragma unroll
      for (int ni = 0; ni < 2; ++ni) {
        int ncol = waveN * 32 + ni * 16 + l16;
        const unsigned short* bp = &sB[buf][ncol * LDB + ks * 32 + hl * 16];
#pragma unroll
        for (int q = 0; q < 8; ++q) bw[ni].u[q] = *(const unsigned*)(bp + 2 * q);
      }
#pragma unroll
      for (int mi = 0; mi < 2; ++mi)
#pragma unroll
        for (int ni = 0; ni < 2; ++ni)
          acc[mi][ni] = __builtin_amdgcn_wmma_f32_16x16x32_bf16(
              false, af[mi].v, false, bw[ni].v, (short)0, acc[mi][ni], false, false);
    }
  };

  // ---- prologue: stage tile 0 ----
  {
    bool asy = stageA(0, 0);
    float breg[16];
    loadB(0, breg);
    storeB(0, breg);
    if (asy) asm volatile("s_wait_asynccnt 0" ::: "memory");
    __syncthreads();
  }

  // ---- pipelined main loop ----
  for (int ti = 0; ti < nTiles; ++ti) {
    const int cb = ti & 1;
    const int nb = cb ^ 1;
    const bool hasNext = (ti + 1 < nTiles);
    float breg[16];
    bool asy = false;
    if (hasNext) {
      asy = stageA(nb, (ti + 1) * BK);   // async issue overlaps with WMMA below
      loadB((ti + 1) * BK, breg);        // global loads in flight during compute
      // keep far-ahead tile warm in L2 (global_prefetch_b8)
      if (ti + 2 < nTiles) {
        int row = tid >> 1;
        __builtin_prefetch(A + (size_t)(m0 + row) * K + (ti + 2) * BK, 0, 1);
      }
    }
    compute(cb);
    if (hasNext) {
      storeB(nb, breg);
      if (asy) asm volatile("s_wait_asynccnt 0" ::: "memory");
      __syncthreads();
    }
  }

  // ---- fused epilogue (C layout per ISA: VGPR r -> M = r + 8*half, N = lane%16)
  const int mBaseW = m0 + waveM * 32;
  const int nBaseW = n0 + waveN * 32;
#pragma unroll
  for (int mi = 0; mi < 2; ++mi) {
#pragma unroll
    for (int ni = 0; ni < 2; ++ni) {
#pragma unroll
      for (int r = 0; r < 8; ++r) {
        int m = mBaseW + mi * 16 + r + hl * 8;
        int n = nBaseW + ni * 16 + l16;
        if (m < M && n < N) {
          float v = acc[mi][ni][r];
          if (flags & F_BN)  v = v * gamma[n] + beta[n];
          if (flags & F_RES) v += bfu2f(residual[(size_t)m * N + n]);
          if (flags & F_RELU) v = v > 0.0f ? v : 0.0f;
          if (flags & F_MASKACC) {
            int bb = m / HW;
            float msk = (float)policy[((size_t)bb * T + t_idx) * 5 + act_idx];
            outF[(size_t)m * N + n] += msk * v;
          } else {
            outH[(size_t)m * N + n] = f2bfu(v);
          }
        }
      }
    }
  }
}

// ---------------------------------------------------------------------------
// Stem: conv1 3x3 (3->32) + BN + ReLU, NCHW f32 in -> NHWC bf16 out
// ---------------------------------------------------------------------------
__global__ void stem_conv(const float* __restrict__ x, const float* __restrict__ w,
                          const float* __restrict__ g, const float* __restrict__ b,
                          unsigned short* __restrict__ out, int Bn) {
  int gid = blockIdx.x * blockDim.x + threadIdx.x;
  int total = Bn * 32 * 32 * 32;
  if (gid >= total) return;
  int c = gid & 31;
  int t = gid >> 5;
  int xx = t & 31; t >>= 5;
  int yy = t & 31; t >>= 5;
  int bb = t;
  float s = 0.0f;
  for (int ci = 0; ci < 3; ++ci)
    for (int dy = 0; dy < 3; ++dy) {
      int y = yy - 1 + dy;
      if (y < 0 || y >= 32) continue;
      for (int dx = 0; dx < 3; ++dx) {
        int xp = xx - 1 + dx;
        if (xp < 0 || xp >= 32) continue;
        s += x[(((size_t)bb * 3 + ci) * 32 + y) * 32 + xp] *
             w[((c * 3 + ci) * 3 + dy) * 3 + dx];
      }
    }
  s = s * g[c] + b[c];
  s = s > 0.0f ? s : 0.0f;
  out[gid] = f2bfu(s);
}

// ---------------------------------------------------------------------------
// im2col (NHWC bf16): out[m][k], k = ci*KH*KH + dy*KH + dx (matches OIHW)
// ---------------------------------------------------------------------------
__global__ void im2col_k(const unsigned short* __restrict__ in,
                         unsigned short* __restrict__ out,
                         int Bn, int H, int Wd, int C, int KH, int stride,
                         int pad, int OH, int OW) {
  long long gid = (long long)blockIdx.x * blockDim.x + threadIdx.x;
  long long K = (long long)KH * KH * C;
  long long total = (long long)Bn * OH * OW * K;
  if (gid >= total) return;
  long long m = gid / K;
  int kidx = (int)(gid % K);
  int ci = kidx / (KH * KH);
  int rr = kidx % (KH * KH);
  int dy = rr / KH, dx = rr % KH;
  int ox = (int)(m % OW);
  long long m2 = m / OW;
  int oy = (int)(m2 % OH);
  int bb = (int)(m2 / OH);
  int y = oy * stride - pad + dy;
  int x = ox * stride - pad + dx;
  unsigned short v = 0;
  if (y >= 0 && y < H && x >= 0 && x < Wd)
    v = in[(((size_t)bb * H + y) * Wd + x) * C + ci];
  out[gid] = v;
}

// ---------------------------------------------------------------------------
// Depthwise kxk conv + BN + ReLU (NHWC bf16, weights f32 [C][k][k])
// ---------------------------------------------------------------------------
__global__ void dwconv_bf16(const unsigned short* __restrict__ in,
                            const float* __restrict__ w,
                            const float* __restrict__ g,
                            const float* __restrict__ b,
                            unsigned short* __restrict__ out,
                            int Bn, int H, int Wd, int C, int k, int stride,
                            int OH, int OW) {
  long long gid = (long long)blockIdx.x * blockDim.x + threadIdx.x;
  long long total = (long long)Bn * OH * OW * C;
  if (gid >= total) return;
  int c = (int)(gid % C);
  long long r = gid / C;
  int ox = (int)(r % OW); r /= OW;
  int oy = (int)(r % OH); r /= OH;
  int bb = (int)r;
  int pad = k >> 1;
  float s = 0.0f;
  for (int dy = 0; dy < k; ++dy) {
    int y = oy * stride - pad + dy;
    if (y < 0 || y >= H) continue;
    for (int dx = 0; dx < k; ++dx) {
      int x = ox * stride - pad + dx;
      if (x < 0 || x >= Wd) continue;
      s += bfu2f(in[(((size_t)bb * H + y) * Wd + x) * C + c]) *
           w[(c * k + dy) * k + dx];
    }
  }
  s = s * g[c] + b[c];
  s = s > 0.0f ? s : 0.0f;
  out[gid] = f2bfu(s);
}

__global__ void fill_zero_f32(float* __restrict__ p, long long n) {
  long long i = (long long)blockIdx.x * blockDim.x + threadIdx.x;
  if (i < n) p[i] = 0.0f;
}

__global__ void cvt_f32_bf16(const float* __restrict__ in,
                             unsigned short* __restrict__ out, long long n) {
  long long i = (long long)blockIdx.x * blockDim.x + threadIdx.x;
  if (i < n) out[i] = f2bfu(in[i]);
}

// ---------------------------------------------------------------------------
// Head: mean-pool over HW + FC (1280 -> 10)
// ---------------------------------------------------------------------------
__global__ void pool_fc(const unsigned short* __restrict__ h,
                        const float* __restrict__ wl,
                        const float* __restrict__ bl,
                        float* __restrict__ logits,
                        int HWs, int C, int NC) {
  __shared__ float pooled[1280];
  int bb = blockIdx.x;
  for (int c = threadIdx.x; c < C; c += blockDim.x) {
    float s = 0.0f;
    for (int sp = 0; sp < HWs; ++sp)
      s += bfu2f(h[((size_t)bb * HWs + sp) * C + c]);
    pooled[c] = s / (float)HWs;
  }
  __syncthreads();
  for (int j = threadIdx.x; j < NC; j += blockDim.x) {
    float s = bl[j];
    for (int c = 0; c < C; ++c) s += pooled[c] * wl[j * C + c];
    logits[bb * NC + j] = s;
  }
}

__global__ void latency_k(const int* __restrict__ pol,
                          const float* __restrict__ lt,
                          float* __restrict__ out, int Bn, int T) {
  int bb = blockIdx.x * blockDim.x + threadIdx.x;
  if (bb >= Bn) return;
  float s = 0.0f;
  for (int t = 0; t < T; ++t)
    for (int a = 0; a < 5; ++a)
      s += (float)pol[((size_t)bb * T + t) * 5 + a] * lt[t * 5 + a];
  out[bb] = s;
}

// ---------------------------------------------------------------------------
// Host orchestration
// ---------------------------------------------------------------------------
struct BranchP {
  const float *w1, *g1, *b1, *w2, *g2, *b2, *w3, *g3, *b3, *ws, *gs, *bs;
};
struct LayerSpec { int inp, oup, st; };

static int build_specs(LayerSpec* sp) {
  const int cfg[7][4] = {{6,16,1,1},{6,24,2,1},{6,32,3,2},{6,64,4,2},
                         {6,96,3,1},{6,160,3,2},{6,320,1,1}};
  int inp = 32, n = 0;
  for (int s = 0; s < 7; ++s)
    for (int r = 0; r < cfg[s][2]; ++r) {
      sp[n].inp = inp; sp[n].oup = cfg[s][1];
      sp[n].st = (r == 0) ? cfg[s][3] : 1;
      inp = cfg[s][1]; ++n;
    }
  return n;  // 17
}

static inline unsigned gblocks(long long total, int tpb) {
  return (unsigned)((total + tpb - 1) / tpb);
}

static void launch_gemm(hipStream_t s, const unsigned short* A, const float* W,
                        float* outF, unsigned short* outH,
                        const float* g, const float* b,
                        const unsigned short* res, const int* pol,
                        int t, int i, int T, int HW,
                        int M, int N, int K, int flags) {
  dim3 grid((N + BN - 1) / BN, (M + BM - 1) / BM);
  wmma_gemm_bf16<<<grid, 256, 0, s>>>(A, W, outF, outH, g, b, res, pol,
                                      t, i, T, HW, M, N, K, flags);
}

extern "C" void kernel_launch(void* const* d_in, const int* in_sizes, int n_in,
                              void* d_out, int out_size, void* d_ws, size_t ws_size,
                              hipStream_t stream) {
  (void)in_sizes; (void)n_in; (void)out_size; (void)ws_size;

  const float* x      = (const float*)d_in[0];
  const int*   policy = (const int*)d_in[1];
  int cur = 2;
  auto nx = [&]() -> const float* { return (const float*)d_in[cur++]; };

  // params pytree, sorted top-level keys: b1,b2,bl,conv1,conv2,g1,g2,layers,wl
  const float* b1s   = nx();
  const float* b2h   = nx();
  const float* bl    = nx();
  const float* conv1 = nx();
  const float* conv2 = nx();
  const float* g1s   = nx();
  const float* g2h   = nx();

  LayerSpec sp[17];
  const int T = build_specs(sp);
  BranchP bp[17][5] = {};
  for (int t = 0; t < T; ++t) {
    int inp = sp[t].inp, oup = sp[t].oup, st = sp[t].st;
    for (int i = 0; i < 5; ++i) {
      BranchP p = {};
      if (i < 4) {  // inverted residual; sorted keys b1,b2,b3,[bs],g1..g3,[gs],w1..w3,[ws]
        bool sc = (st == 1 && inp != oup);
        p.b1 = nx(); p.b2 = nx(); p.b3 = nx(); if (sc) p.bs = nx();
        p.g1 = nx(); p.g2 = nx(); p.g3 = nx(); if (sc) p.gs = nx();
        p.w1 = nx(); p.w2 = nx(); p.w3 = nx(); if (sc) p.ws = nx();
      } else {      // basic block; sorted keys b1,b2,[bs],g1,g2,[gs],w1,w2,[ws]
        bool sc = (st != 1 || inp != oup);
        p.b1 = nx(); p.b2 = nx(); if (sc) p.bs = nx();
        p.g1 = nx(); p.g2 = nx(); if (sc) p.gs = nx();
        p.w1 = nx(); p.w2 = nx(); if (sc) p.ws = nx();
      }
      bp[t][i] = p;
    }
  }
  const float* wl  = nx();
  const float* lat = nx();

  // ---- workspace carve (bf16 activations / patches, f32 accumulator) ----
  char* wsp = (char*)d_ws;
  auto alloc = [&](size_t bytes) -> void* {
    void* p = (void*)wsp;
    wsp += (bytes + 255) & ~(size_t)255;
    return p;
  };
  unsigned short* bufCur = (unsigned short*)alloc((size_t)5  * 1024 * 1024 * 2);
  unsigned short* bufT1  = (unsigned short*)alloc((size_t)19 * 1024 * 1024 * 2);
  unsigned short* bufT2  = (unsigned short*)alloc((size_t)19 * 1024 * 1024 * 2);
  unsigned short* bufSC  = (unsigned short*)alloc((size_t)4  * 1024 * 1024 * 2);
  unsigned short* bufPat = (unsigned short*)alloc((size_t)29 * 1024 * 1024 * 2);
  float*          bufAcc = (float*)alloc((size_t)4 * 1024 * 1024 * 4);

  const int Bn = 128;

  // ---- stem ----
  {
    long long total = (long long)Bn * 32 * 32 * 32;
    stem_conv<<<gblocks(total, 256), 256, 0, stream>>>(x, conv1, g1s, b1s, bufCur, Bn);
  }

  // ---- 17 masked super-layers ----
  int Hs = 32;
  for (int t = 0; t < T; ++t) {
    const int inp = sp[t].inp, oup = sp[t].oup, st = sp[t].st;
    const int OH = Hs / st;
    const int Min  = Bn * Hs * Hs;
    const int Mout = Bn * OH * OH;
    const int HW = OH * OH;

    fill_zero_f32<<<gblocks((long long)Mout * oup, 256), 256, 0, stream>>>(
        bufAcc, (long long)Mout * oup);

    for (int i = 0; i < 5; ++i) {
      const BranchP& p = bp[t][i];
      if (i < 4) {
        const int kk = (i < 2) ? 3 : 5;
        const int e  = (i & 1) ? 3 : 6;
        const int pl = e * inp;
        // expand 1x1 + BN + ReLU
        launch_gemm(stream, bufCur, p.w1, nullptr, bufT1, p.g1, p.b1,
                    nullptr, nullptr, 0, 0, 1, 1, Min, pl, inp, F_BN | F_RELU);
        // depthwise kxk stride st + BN + ReLU
        dwconv_bf16<<<gblocks((long long)Mout * pl, 256), 256, 0, stream>>>(
            bufT1, p.w2, p.g2, p.b2, bufT2, Bn, Hs, Hs, pl, kk, st, OH, OH);
        // shortcut (stride 1 only)
        const unsigned short* res = nullptr;
        if (st == 1) {
          if (p.ws) {
            launch_gemm(stream, bufCur, p.ws, nullptr, bufSC, p.gs, p.bs,
                        nullptr, nullptr, 0, 0, 1, 1, Min, oup, inp, F_BN);
            res = bufSC;
          } else {
            res = bufCur;  // identity (inp == oup)
          }
        }
        // project 1x1 + BN (+ residual), policy-masked accumulate
        launch_gemm(stream, bufT2, p.w3, bufAcc, nullptr, p.g3, p.b3,
                    res, policy, t, i, T, HW, Mout, oup, pl,
                    F_BN | F_MASKACC | (res ? F_RES : 0));
      } else {
        // basic block: conv3x3(st) + BN + ReLU
        im2col_k<<<gblocks((long long)Mout * inp * 9, 256), 256, 0, stream>>>(
            bufCur, bufPat, Bn, Hs, Hs, inp, 3, st, 1, OH, OH);
        launch_gemm(stream, bufPat, p.w1, nullptr, bufT1, p.g1, p.b1,
                    nullptr, nullptr, 0, 0, 1, 1, Mout, oup, inp * 9, F_BN | F_RELU);
        // conv3x3(1) patch
        im2col_k<<<gblocks((long long)Mout * oup * 9, 256), 256, 0, stream>>>(
            bufT1, bufPat, Bn, OH, OH, oup, 3, 1, 1, OH, OH);
        // shortcut
        const unsigned short* res;
        if (p.ws) {
          const unsigned short* scA = bufCur;
          if (st != 1) {
            im2col_k<<<gblocks((long long)Mout * inp, 256), 256, 0, stream>>>(
                bufCur, bufT2, Bn, Hs, Hs, inp, 1, st, 0, OH, OH);
            scA = bufT2;
          }
          launch_gemm(stream, scA, p.ws, nullptr, bufSC, p.gs, p.bs,
                      nullptr, nullptr, 0, 0, 1, 1, Mout, oup, inp, F_BN);
          res = bufSC;
        } else {
          res = bufCur;  // identity
        }
        // conv2 + BN + residual + ReLU, policy-masked accumulate
        launch_gemm(stream, bufPat, p.w2, bufAcc, nullptr, p.g2, p.b2,
                    res, policy, t, i, T, HW, Mout, oup, oup * 9,
                    F_BN | F_RES | F_RELU | F_MASKACC);
      }
    }

    // finalize: f32 accumulator -> bf16 activations for next layer
    cvt_f32_bf16<<<gblocks((long long)Mout * oup, 256), 256, 0, stream>>>(
        bufAcc, bufCur, (long long)Mout * oup);
    Hs = OH;
  }

  // ---- head: conv2 1x1 (320->1280) + BN + ReLU ----
  launch_gemm(stream, bufCur, conv2, nullptr, bufT1, g2h, b2h,
              nullptr, nullptr, 0, 0, 1, 1, Bn * Hs * Hs, 1280, 320,
              F_BN | F_RELU);
  // mean-pool + FC -> logits
  pool_fc<<<Bn, 256, 0, stream>>>(bufT1, wl, bl, (float*)d_out,
                                  Hs * Hs, 1280, 10);
  // latency einsum -> second output
  latency_k<<<1, 128, 0, stream>>>(policy, lat, (float*)d_out + Bn * 10, Bn, T);
}
